// TransformerEncoderLiquidNetworkModel_52922587022078
// MI455X (gfx1250) — matrix operand order repaired and checked by reading, not compiled, not run on hardware
//
#include <hip/hip_runtime.h>
#include <hip/hip_bf16.h>
#include <math.h>

// Model dims
#define BB   32
#define SS   512
#define INN  128
#define HH   512
#define NHH  8
#define HD   64
#define FFF  2048
#define NLL  2
#define MM   (BB * SS)          // 16384 rows
#define DTT  0.1f
#define EPSS 1e-5f

typedef __bf16 bf16;
typedef bf16  bf16x4  __attribute__((ext_vector_type(4)));
typedef bf16  bf16x8  __attribute__((ext_vector_type(8)));
typedef bf16  bf16x16 __attribute__((ext_vector_type(16)));
typedef float f32x4v  __attribute__((ext_vector_type(4)));
typedef float f32x8   __attribute__((ext_vector_type(8)));

__device__ __forceinline__ f32x8 wmma_bf16(bf16x16 a, bf16x16 b, f32x8 c) {
  return __builtin_amdgcn_wmma_f32_16x16x32_bf16(false, a, false, b, (short)0, c,
                                                 false, false);
}

// Build A-fragment (16x32 bf16 layout) from two contiguous 8-element halves:
// lane half h reads K = h*16 + 0..7 (lo) and K = h*16 + 8..15 (hi).
__device__ __forceinline__ bf16x16 make_afrag(bf16x8 lo, bf16x8 hi) {
  return __builtin_shufflevector(lo, hi, 0, 1, 2, 3, 4, 5, 6, 7, 8, 9, 10, 11,
                                 12, 13, 14, 15);
}

// ---------------------------------------------------------------------------
// Elementwise f32 -> bf16 (4 per thread, vectorized)
// ---------------------------------------------------------------------------
__global__ __launch_bounds__(256) void cvt_bf16(const float* __restrict__ s,
                                                bf16* __restrict__ d, int n4) {
  const int i = blockIdx.x * 256 + threadIdx.x;
  if (i < n4) {
    f32x4v v = ((const f32x4v*)s)[i];
    bf16x4 o;
    o[0] = (bf16)v[0]; o[1] = (bf16)v[1]; o[2] = (bf16)v[2]; o[3] = (bf16)v[3];
    ((bf16x4*)d)[i] = o;
  }
}

// Transpose-convert rec_w [512,512]: d[n*512+k] = (bf16)s[k*512+n]
__global__ __launch_bounds__(256) void cvt_bf16_t(const float* __restrict__ s,
                                                  bf16* __restrict__ d) {
  const int i = blockIdx.x * 256 + threadIdx.x;  // 262144 threads
  const int k = i >> 9, n = i & 511;
  d[(size_t)n * HH + k] = (bf16)s[i];
}

// ---------------------------------------------------------------------------
// D[M,N] = act( A[M,K](bf16) @ W[N,K](bf16)^T + bias[N] ).
// 128x64 block, 8 waves x (16x64). Optional f32 out, bf16 out, V-transpose out.
// ---------------------------------------------------------------------------
__global__ __launch_bounds__(256) void gemm_bf16(
    const bf16* __restrict__ A, const bf16* __restrict__ W,
    const float* __restrict__ bias, float* __restrict__ Df,
    bf16* __restrict__ Db, bf16* __restrict__ vT, int M, int N, int K,
    int relu) {
  const int tid  = threadIdx.x;
  const int wave = tid >> 5, lane = tid & 31;
  const int half = lane >> 4, l16 = lane & 15;
  const int bm = blockIdx.y * 128 + wave * 16;
  const int bn = blockIdx.x * 64;

  f32x8 acc[4] = {};
  const bf16* arow = A + (size_t)(bm + l16) * K;
  const bf16* w0 = W + (size_t)(bn + 0 * 16 + l16) * K + half * 16;
  const bf16* w1 = W + (size_t)(bn + 1 * 16 + l16) * K + half * 16;
  const bf16* w2 = W + (size_t)(bn + 2 * 16 + l16) * K + half * 16;
  const bf16* w3 = W + (size_t)(bn + 3 * 16 + l16) * K + half * 16;

  for (int k0 = 0; k0 < K; k0 += 32) {
    // Issue all fragment loads up front (distinct regs -> MLP)
    bf16x8  alo = *(const bf16x8*)(arow + k0 + half * 16);
    bf16x8  ahi = *(const bf16x8*)(arow + k0 + 8 + half * 16);
    bf16x16 b0  = *(const bf16x16*)(w0 + k0);
    bf16x16 b1  = *(const bf16x16*)(w1 + k0);
    bf16x16 b2  = *(const bf16x16*)(w2 + k0);
    bf16x16 b3  = *(const bf16x16*)(w3 + k0);
    __builtin_prefetch((const void*)(arow + k0 + 64), 0, 1);
    bf16x16 af = make_afrag(alo, ahi);
    acc[0] = wmma_bf16(af, b0, acc[0]);
    acc[1] = wmma_bf16(af, b1, acc[1]);
    acc[2] = wmma_bf16(af, b2, acc[2]);
    acc[3] = wmma_bf16(af, b3, acc[3]);
  }
#pragma unroll
  for (int nt = 0; nt < 4; ++nt) {
    const int col = bn + nt * 16 + l16;
    const float bv = bias[col];
#pragma unroll
    for (int r = 0; r < 8; ++r) {
      const int row = bm + r + 8 * half;
      float v = acc[nt][r] + bv;
      if (relu) v = fmaxf(v, 0.f);
      if (Df) Df[(size_t)row * N + col] = v;
      if (Db) Db[(size_t)row * N + col] = (bf16)v;
      if (vT && col >= 2 * HH) {  // scatter V transposed: vT[b,h,d,s]
        const int c2 = col - 2 * HH;
        const int hh = c2 >> 6, d = c2 & 63;
        const int bb = row >> 9, s = row & 511;
        vT[(((size_t)bb * NHH + hh) * HD + d) * SS + s] = (bf16)v;
      }
    }
  }
}

// ---------------------------------------------------------------------------
// Fused attention for one (batch, head, 64-row q block).
// LDS: scores 64x512 f32 (128KB) | probs 64x512 bf16 (64KB) | red 256 f32.
// ---------------------------------------------------------------------------
__global__ __launch_bounds__(256) void attn_fused(const bf16* __restrict__ qkv,
                                                  const bf16* __restrict__ vT,
                                                  bf16* __restrict__ ctx) {
  extern __shared__ char smem[];
  float* sc  = (float*)smem;                                // 64*512 f32
  bf16*  pb  = (bf16*)(smem + 64 * 512 * 4);                // 64*512 bf16
  float* red = (float*)(smem + 64 * 512 * 4 + 64 * 512 * 2);

  const int b = blockIdx.z, hh = blockIdx.y, qb = blockIdx.x * 64;
  const int tid  = threadIdx.x;
  const int wave = tid >> 5, lane = tid & 31;
  const int half = lane >> 4, l16 = lane & 15;
  const int QKV = 3 * HH;

  // Phase 1: scores[64,512] = q @ k^T * (1/8)
  {
    const int m_sub  = wave & 3;
    const int n_base = (wave >> 2) * 256;
    const bf16* qrow =
        qkv + (size_t)(b * SS + qb + m_sub * 16 + l16) * QKV + hh * HD;
    bf16x16 a0 = make_afrag(*(const bf16x8*)(qrow + half * 16),
                            *(const bf16x8*)(qrow + 8 + half * 16));
    bf16x16 a1 = make_afrag(*(const bf16x8*)(qrow + 32 + half * 16),
                            *(const bf16x8*)(qrow + 40 + half * 16));
    for (int nt = 0; nt < 16; ++nt) {
      const int ncol = n_base + nt * 16;
      const bf16* krow =
          qkv + (size_t)(b * SS + ncol + l16) * QKV + HH + hh * HD + half * 16;
      bf16x16 b0 = *(const bf16x16*)(krow);
      bf16x16 b1 = *(const bf16x16*)(krow + 32);
      f32x8 acc = {};
      acc = wmma_bf16(a0, b0, acc);
      acc = wmma_bf16(a1, b1, acc);
#pragma unroll
      for (int r = 0; r < 8; ++r)
        sc[(m_sub * 16 + r + 8 * half) * SS + ncol + l16] = acc[r] * 0.125f;
    }
  }
  __syncthreads();

  // Phase 2: row softmax (4 threads per row), probs -> bf16
  {
    const int row = tid >> 2, q4 = tid & 3;
    float* rp = sc + row * SS + q4 * 128;
    float mx = -3.0e38f;
    for (int i = 0; i < 128; ++i) mx = fmaxf(mx, rp[i]);
    red[row * 4 + q4] = mx;
    __syncthreads();
    const float rm = fmaxf(fmaxf(red[row * 4 + 0], red[row * 4 + 1]),
                           fmaxf(red[row * 4 + 2], red[row * 4 + 3]));
    __syncthreads();
    float s = 0.f;
    for (int i = 0; i < 128; ++i) {
      const float e = __expf(rp[i] - rm);
      rp[i] = e;
      s += e;
    }
    red[row * 4 + q4] = s;
    __syncthreads();
    const float inv = 1.f / (red[row * 4 + 0] + red[row * 4 + 1] +
                             red[row * 4 + 2] + red[row * 4 + 3]);
    bf16* pp = pb + row * SS + q4 * 128;
    for (int i = 0; i < 128; ++i) pp[i] = (bf16)(rp[i] * inv);
  }
  __syncthreads();

  // Phase 3: ctx[64,64] = P[64,512] @ V[512,64], V from vT[b,h,d,s]
  {
    const int m_sub = wave & 3;
    const int n_off = (wave >> 2) * 32;
    const bf16* prow = pb + (m_sub * 16 + l16) * SS;
    const bf16* v0 =
        vT + (((size_t)b * NHH + hh) * HD + n_off + 0 * 16 + l16) * SS +
        half * 16;
    const bf16* v1 =
        vT + (((size_t)b * NHH + hh) * HD + n_off + 1 * 16 + l16) * SS +
        half * 16;
    f32x8 acc[2] = {};
    for (int k0 = 0; k0 < SS; k0 += 32) {
      bf16x8  plo = *(const bf16x8*)(prow + k0 + half * 16);
      bf16x8  phi = *(const bf16x8*)(prow + k0 + 8 + half * 16);
      bf16x16 bb0 = *(const bf16x16*)(v0 + k0);
      bf16x16 bb1 = *(const bf16x16*)(v1 + k0);
      bf16x16 af  = make_afrag(plo, phi);
      acc[0] = wmma_bf16(af, bb0, acc[0]);
      acc[1] = wmma_bf16(af, bb1, acc[1]);
    }
#pragma unroll
    for (int nt = 0; nt < 2; ++nt) {
      const int col = hh * HD + n_off + nt * 16 + l16;
#pragma unroll
      for (int r = 0; r < 8; ++r) {
        const int row = qb + m_sub * 16 + r + 8 * half;
        ctx[(size_t)(b * SS + row) * HH + col] = (bf16)acc[nt][r];
      }
    }
  }
}

// ---------------------------------------------------------------------------
// out = LayerNorm(x + y) * g + b   (one block per row), f32 + bf16 mirror
// ---------------------------------------------------------------------------
__global__ __launch_bounds__(256) void add_ln(const float* __restrict__ x,
                                              const float* __restrict__ y,
                                              const float* __restrict__ g,
                                              const float* __restrict__ bta,
                                              float* __restrict__ out,
                                              bf16* __restrict__ outb) {
  __shared__ float s1[256], s2[256];
  const int row = blockIdx.x, tid = threadIdx.x;
  const float* xr = x + (size_t)row * HH;
  const float* yr = y + (size_t)row * HH;
  const float v0 = xr[tid] + yr[tid];
  const float v1 = xr[tid + 256] + yr[tid + 256];
  s1[tid] = v0 + v1;
  s2[tid] = v0 * v0 + v1 * v1;
  __syncthreads();
  for (int st = 128; st > 0; st >>= 1) {
    if (tid < st) { s1[tid] += s1[tid + st]; s2[tid] += s2[tid + st]; }
    __syncthreads();
  }
  const float mu   = s1[0] * (1.f / HH);
  const float var  = s2[0] * (1.f / HH) - mu * mu;
  const float rstd = rsqrtf(var + EPSS);
  const float o0 = (v0 - mu) * rstd * g[tid] + bta[tid];
  const float o1 = (v1 - mu) * rstd * g[tid + 256] + bta[tid + 256];
  float* orow = out + (size_t)row * HH;
  orow[tid]       = o0;
  orow[tid + 256] = o1;
  bf16* brow = outb + (size_t)row * HH;
  brow[tid]       = (bf16)o0;
  brow[tid + 256] = (bf16)o1;
}

// ---------------------------------------------------------------------------
// Sequential liquid ODE scan: single workgroup (32 waves), hid state in LDS
// (f32 master + bf16 mirror), per-step [32,512]x[512,512] matvec via WMMA.
// rec_wT is pre-transposed bf16 [N,K] so B-fragments are contiguous.
// ---------------------------------------------------------------------------
__global__ __launch_bounds__(1024) void liquid_scan(
    const float* __restrict__ inp, const bf16* __restrict__ rec_wT,
    const float* __restrict__ tau, float* __restrict__ hT) {
  extern __shared__ char smem[];
  float* hidf = (float*)smem;                 // 32*512 f32 (64KB)
  bf16*  hidb = (bf16*)(smem + BB * HH * 4);  // 32*512 bf16 (32KB)
  const int tid = threadIdx.x;
  for (int i = tid; i < BB * HH; i += 1024) {
    hidf[i] = 0.f;
    hidb[i] = (bf16)0.f;
  }
  __syncthreads();

  const int wave = tid >> 5, lane = tid & 31;
  const int half = lane >> 4, l16 = lane & 15;
  const int m_sub = wave >> 4;         // 0..1  (batch-row tile)
  const int ncol  = (wave & 15) * 32;  // 0..480 (output cols)
  const bf16* hrow = hidb + (m_sub * 16 + l16) * HH;
  const bf16* r0 = rec_wT + (size_t)(ncol + 0 * 16 + l16) * HH + half * 16;
  const bf16* r1 = rec_wT + (size_t)(ncol + 1 * 16 + l16) * HH + half * 16;

  for (int t = 0; t < SS; ++t) {
    f32x8 acc[2] = {};
    for (int k0 = 0; k0 < HH; k0 += 32) {
      bf16x8  alo = *(const bf16x8*)(hrow + k0 + half * 16);
      bf16x8  ahi = *(const bf16x8*)(hrow + k0 + 8 + half * 16);
      bf16x16 bb0 = *(const bf16x16*)(r0 + k0);
      bf16x16 bb1 = *(const bf16x16*)(r1 + k0);
      bf16x16 af  = make_afrag(alo, ahi);
      acc[0] = wmma_bf16(af, bb0, acc[0]);
      acc[1] = wmma_bf16(af, bb1, acc[1]);
    }
    __syncthreads();  // all hidb reads for this step complete
#pragma unroll
    for (int nt = 0; nt < 2; ++nt) {
      const int col = ncol + nt * 16 + l16;
      const float tv = tau[col];
#pragma unroll
      for (int r = 0; r < 8; ++r) {
        const int brow = m_sub * 16 + r + 8 * half;
        const float hv = hidf[brow * HH + col];
        const float xt = inp[((size_t)brow * SS + t) * HH + col];
        const float f  = tanhf(xt + acc[nt][r]);
        float nv = hv + (-hv / tv + f) * DTT;
        nv = fminf(fmaxf(nv, -10.f), 10.f);
        hidf[brow * HH + col] = nv;
        hidb[brow * HH + col] = (bf16)nv;
      }
    }
    __syncthreads();  // updates visible before next step's WMMA reads
  }
  for (int i = tid; i < BB * HH; i += 1024) hT[i] = hidf[i];
}

// ---------------------------------------------------------------------------
// out[b] = dot(hT[b,:], fc_w) + fc_b[0]
// ---------------------------------------------------------------------------
__global__ __launch_bounds__(256) void fc_out(const float* __restrict__ hT,
                                              const float* __restrict__ w,
                                              const float* __restrict__ bias,
                                              float* __restrict__ out) {
  __shared__ float s1[256];
  const int row = blockIdx.x, tid = threadIdx.x;
  s1[tid] = hT[(size_t)row * HH + tid] * w[tid] +
            hT[(size_t)row * HH + tid + 256] * w[tid + 256];
  __syncthreads();
  for (int st = 128; st > 0; st >>= 1) {
    if (tid < st) s1[tid] += s1[tid + st];
    __syncthreads();
  }
  if (tid == 0) out[row] = s1[0] + bias[0];
}

// ---------------------------------------------------------------------------
extern "C" void kernel_launch(void* const* d_in, const int* in_sizes, int n_in,
                              void* d_out, int out_size, void* d_ws,
                              size_t ws_size, hipStream_t stream) {
  (void)in_sizes; (void)n_in; (void)out_size; (void)ws_size;
  const float* x      = (const float*)d_in[0];
  const float* w_proj = (const float*)d_in[1];
  const float* b_proj = (const float*)d_in[2];
  const float* in_w   = (const float*)d_in[3];
  const float* in_b   = (const float*)d_in[4];
  const float* out_w  = (const float*)d_in[5];
  const float* out_b  = (const float*)d_in[6];
  const float* ln1_g  = (const float*)d_in[7];
  const float* ln1_b  = (const float*)d_in[8];
  const float* ln2_g  = (const float*)d_in[9];
  const float* ln2_b  = (const float*)d_in[10];
  const float* ff_w1  = (const float*)d_in[11];
  const float* ff_b1  = (const float*)d_in[12];
  const float* ff_w2  = (const float*)d_in[13];
  const float* ff_b2  = (const float*)d_in[14];
  const float* liq_w  = (const float*)d_in[15];
  const float* liq_b  = (const float*)d_in[16];
  const float* tau    = (const float*)d_in[17];
  const float* rec_w  = (const float*)d_in[18];
  const float* fc_w   = (const float*)d_in[19];
  const float* fc_b   = (const float*)d_in[20];
  float* out = (float*)d_out;

  // ---- workspace carve-up (byte offsets, all 256B-aligned) ----
  char* ws = (char*)d_ws;
  float* h_f32  = (float*)(ws);                       // 33554432 B
  float* c_f32  = (float*)(ws + 33554432ull);         // 33554432 B
  bf16*  h_bf   = (bf16*)(ws + 67108864ull);          // 16777216 B
  bf16*  sc_bf  = (bf16*)(ws + 83886080ull);          // 67108864 B (qkv/ff1)
  bf16*  ctx_bf = (bf16*)(ws + 150994944ull);         // 16777216 B
  bf16*  vT_bf  = (bf16*)(ws + 167772160ull);         // 16777216 B
  bf16*  x_bf   = (bf16*)(ws + 184549376ull);         //  4194304 B
  bf16*  wts    = (bf16*)(ws + 188743680ull);         // 13762560 B
  float* hT     = (float*)(ws + 202506240ull);        //    65536 B

  // bf16 weight arena sub-offsets (elements)
  bf16* wproj_bf = wts;                       // 65536
  bf16* inw_bf   = wproj_bf + 65536;          // 1572864
  bf16* outw_bf  = inw_bf + 1572864;          // 524288
  bf16* ffw1_bf  = outw_bf + 524288;          // 2097152
  bf16* ffw2_bf  = ffw1_bf + 2097152;         // 2097152
  bf16* liqw_bf  = ffw2_bf + 2097152;         // 262144
  bf16* recT_bf  = liqw_bf + 262144;          // 262144

  const size_t attn_smem = (size_t)64 * 512 * 4 + (size_t)64 * 512 * 2 + 1024;
  const size_t liq_smem  = (size_t)BB * HH * 4 + (size_t)BB * HH * 2;
  (void)hipFuncSetAttribute((const void*)attn_fused,
                            hipFuncAttributeMaxDynamicSharedMemorySize,
                            (int)attn_smem);
  (void)hipFuncSetAttribute((const void*)liquid_scan,
                            hipFuncAttributeMaxDynamicSharedMemorySize,
                            (int)liq_smem);

  dim3 blk(256);
  // ---- one-time (per launch) bf16 conversions ----
  cvt_bf16<<<dim3((MM * INN / 4 + 255) / 256), blk, 0, stream>>>(x, x_bf,
                                                                 MM * INN / 4);
  cvt_bf16<<<dim3(64), blk, 0, stream>>>(w_proj, wproj_bf, 65536 / 4);
  cvt_bf16<<<dim3(1536), blk, 0, stream>>>(in_w, inw_bf, 1572864 / 4);
  cvt_bf16<<<dim3(512), blk, 0, stream>>>(out_w, outw_bf, 524288 / 4);
  cvt_bf16<<<dim3(2048), blk, 0, stream>>>(ff_w1, ffw1_bf, 2097152 / 4);
  cvt_bf16<<<dim3(2048), blk, 0, stream>>>(ff_w2, ffw2_bf, 2097152 / 4);
  cvt_bf16<<<dim3(256), blk, 0, stream>>>(liq_w, liqw_bf, 262144 / 4);
  cvt_bf16_t<<<dim3(1024), blk, 0, stream>>>(rec_w, recT_bf);

  // ---- h = x @ w_proj^T + b_proj (f32 + bf16) ----
  gemm_bf16<<<dim3(HH / 64, MM / 128), blk, 0, stream>>>(
      x_bf, wproj_bf, b_proj, h_f32, h_bf, nullptr, MM, HH, INN, 0);

  for (int l = 0; l < NLL; ++l) {
    // qkv (bf16 only) + transposed V scatter
    gemm_bf16<<<dim3(3 * HH / 64, MM / 128), blk, 0, stream>>>(
        h_bf, inw_bf + (size_t)l * 3 * HH * HH, in_b + (size_t)l * 3 * HH,
        nullptr, sc_bf, vT_bf, MM, 3 * HH, HH, 0);
    // fused attention -> ctx (bf16)
    attn_fused<<<dim3(SS / 64, NHH, BB), blk, attn_smem, stream>>>(
        sc_bf, vT_bf, ctx_bf);
    // out projection (f32 only)
    gemm_bf16<<<dim3(HH / 64, MM / 128), blk, 0, stream>>>(
        ctx_bf, outw_bf + (size_t)l * HH * HH, out_b + (size_t)l * HH, c_f32,
        nullptr, nullptr, MM, HH, HH, 0);
    // x = LN(x + attn_out)  (f32 + bf16)
    add_ln<<<dim3(MM), blk, 0, stream>>>(h_f32, c_f32, ln1_g + (size_t)l * HH,
                                         ln1_b + (size_t)l * HH, h_f32, h_bf);
    // ff1 = relu(x @ w1^T + b1)  (bf16 only)
    gemm_bf16<<<dim3(FFF / 64, MM / 128), blk, 0, stream>>>(
        h_bf, ffw1_bf + (size_t)l * FFF * HH, ff_b1 + (size_t)l * FFF, nullptr,
        sc_bf, nullptr, MM, FFF, HH, 1);
    // ff2 = ff1 @ w2^T + b2  (f32 only)
    gemm_bf16<<<dim3(HH / 64, MM / 128), blk, 0, stream>>>(
        sc_bf, ffw2_bf + (size_t)l * HH * FFF, ff_b2 + (size_t)l * HH, c_f32,
        nullptr, nullptr, MM, HH, FFF, 0);
    // x = LN(x + ff2)  (f32 + bf16)
    add_ln<<<dim3(MM), blk, 0, stream>>>(h_f32, c_f32, ln2_g + (size_t)l * HH,
                                         ln2_b + (size_t)l * HH, h_f32, h_bf);
  }
  // inp_proj = h @ liq_w^T + liq_b  (f32 only)
  gemm_bf16<<<dim3(HH / 64, MM / 128), blk, 0, stream>>>(
      h_bf, liqw_bf, liq_b, c_f32, nullptr, nullptr, MM, HH, HH, 0);
  // sequential liquid recurrence (single persistent workgroup)
  liquid_scan<<<dim3(1), dim3(1024), liq_smem, stream>>>(c_f32, recT_bf, tau,
                                                         hT);
  // final fc
  fc_out<<<dim3(BB), blk, 0, stream>>>(hT, fc_w, fc_b, out);
}
